// MMSingleStreamBlockRef_15324443312831
// MI455X (gfx1250) — matrix operand order repaired and checked by reading, not compiled, not run on hardware
//
#include <hip/hip_runtime.h>
#include <hip/hip_bf16.h>
#include <math.h>

typedef __bf16 bf16_t;
typedef __attribute__((ext_vector_type(16))) __bf16 v16bf;
typedef __attribute__((ext_vector_type(8)))  float  v8f;

#define D_MODEL 3072
#define NHEAD   24
#define HD      128
#define IMG     2048
#define REFN    512
#define TXT     256
#define LTOT    2816          // IMG + REFN + TXT
#define XLEN    2304          // IMG + TXT
#define MLPD    12288
#define N1      21504         // 3*D + MLP
#define N2CAT   15360         // D + MLP
#define LN_EPS  1e-6f
#define QK_SCALE 0.08838834764831845f  // 1/sqrt(128)

#define BM 256
#define BN 128
#define BK 32
#define KT 64

__device__ inline float gelu_tanh(float x) {
    float x3 = x * x * x;
    return 0.5f * x * (1.0f + tanhf(0.7978845608028654f * (x + 0.044715f * x3)));
}

template <int NT>
__device__ inline float blockReduceSum(float v, float* sbuf) {
    const int tid = threadIdx.x;
    sbuf[tid] = v;
    __syncthreads();
#pragma unroll
    for (int s = NT / 2; s > 0; s >>= 1) {
        if (tid < s) sbuf[tid] += sbuf[tid + s];
        __syncthreads();
    }
    const float r = sbuf[0];
    __syncthreads();
    return r;
}

// ---------------------------------------------------------------------------
// mod = silu(vec) @ mod_w + mod_b            (1 x 3072) x (3072 x 9216)
// ---------------------------------------------------------------------------
__global__ __launch_bounds__(256) void mod_gemv(const float* __restrict__ vec,
                                                const float* __restrict__ W,
                                                const float* __restrict__ b,
                                                float* __restrict__ mod) {
    __shared__ float sv[D_MODEL];
    const int tid = threadIdx.x;
    for (int i = tid; i < D_MODEL; i += 256) {
        const float v = vec[i];
        sv[i] = v / (1.0f + __expf(-v));
    }
    __syncthreads();
    const int j = blockIdx.x * 256 + tid;
    float acc = b[j];
    for (int i = 0; i < D_MODEL; ++i) acc += sv[i] * W[(size_t)i * (3 * D_MODEL) + j];
    mod[j] = acc;
}

// ---------------------------------------------------------------------------
// x_mod = LN(x) * (1 + scale) + shift, scattered into concatenated xm buffer
// ---------------------------------------------------------------------------
__global__ __launch_bounds__(256) void ln_mod_x(const float* __restrict__ x,
                                                const float* __restrict__ mod,
                                                float* __restrict__ xm) {
    const int row = blockIdx.x;   // 0..XLEN-1
    const int tid = threadIdx.x;
    __shared__ float red[256];
    float vals[12];
    float s = 0.f;
#pragma unroll
    for (int i = 0; i < 12; ++i) {
        vals[i] = x[(size_t)row * D_MODEL + tid + i * 256];
        s += vals[i];
    }
    const float mean = blockReduceSum<256>(s, red) * (1.0f / D_MODEL);
    float s2 = 0.f;
#pragma unroll
    for (int i = 0; i < 12; ++i) { const float z = vals[i] - mean; s2 += z * z; }
    const float var  = blockReduceSum<256>(s2, red) * (1.0f / D_MODEL);
    const float rstd = rsqrtf(var + LN_EPS);
    const int drow = (row < IMG) ? row : row + REFN;
#pragma unroll
    for (int i = 0; i < 12; ++i) {
        const int c = tid + i * 256;
        const float y = (vals[i] - mean) * rstd;
        xm[(size_t)drow * D_MODEL + c] = y * (1.0f + mod[D_MODEL + c]) + mod[c];
    }
}

// ---------------------------------------------------------------------------
// in-place LN with affine (ref_norm) on 512 rows
// ---------------------------------------------------------------------------
__global__ __launch_bounds__(256) void ln_affine(float* __restrict__ z,
                                                 const float* __restrict__ w,
                                                 const float* __restrict__ b) {
    const int row = blockIdx.x;
    const int tid = threadIdx.x;
    __shared__ float red[256];
    float vals[12];
    float s = 0.f;
#pragma unroll
    for (int i = 0; i < 12; ++i) {
        vals[i] = z[(size_t)row * D_MODEL + tid + i * 256];
        s += vals[i];
    }
    const float mean = blockReduceSum<256>(s, red) * (1.0f / D_MODEL);
    float s2 = 0.f;
#pragma unroll
    for (int i = 0; i < 12; ++i) { const float d = vals[i] - mean; s2 += d * d; }
    const float var  = blockReduceSum<256>(s2, red) * (1.0f / D_MODEL);
    const float rstd = rsqrtf(var + LN_EPS);
#pragma unroll
    for (int i = 0; i < 12; ++i) {
        const int c = tid + i * 256;
        z[(size_t)row * D_MODEL + c] = (vals[i] - mean) * rstd * w[c] + b[c];
    }
}

// ---------------------------------------------------------------------------
// Tiled GEMM: C = A(MxK) @ B(KxN), bf16 WMMA, f32 accumulate.
// Block tile 256x128, 8 waves each computing 64x64 (4x4 WMMA subtiles).
// B is stored TRANSPOSED in LDS so every fragment load is 2x ds_load_b128.
// EP: 1 = bias, 2 = bias+gelu, 3 = bias then out = resid + C*gate
// ---------------------------------------------------------------------------
template <int EP>
__global__ __launch_bounds__(256) void gemm_bf16(const float* __restrict__ A,
                                                 const float* __restrict__ B,
                                                 const float* __restrict__ bias,
                                                 float* __restrict__ C,
                                                 int M, int N, int K,
                                                 const float* __restrict__ resid,
                                                 const float* __restrict__ gate) {
    __shared__ bf16_t As[BM][BK + 8];   // [m][k]  row stride 80B (16B aligned)
    __shared__ bf16_t Bt[BN][BK + 8];   // [n][k]  transposed
    const int tid  = threadIdx.x;
    const int lane = tid & 31;
    const int wid  = tid >> 5;          // 8 waves
    const int wm   = wid >> 1;          // 0..3 (64-row band)
    const int wn   = wid & 1;           // 0..1 (64-col band)
    const int brow = blockIdx.y * BM;
    const int bcol = blockIdx.x * BN;
    const int ml   = lane & 15;
    const int hh   = lane >> 4;
    const int kb   = hh * 8;            // table-order K base (unused)
    const int kb2  = hh * 16;           // contiguous K base (A and B)

    v8f acc[4][4];
#pragma unroll
    for (int i = 0; i < 4; ++i)
#pragma unroll
        for (int j = 0; j < 4; ++j)
#pragma unroll
            for (int e = 0; e < 8; ++e) acc[i][j][e] = 0.0f;

    const int nk = K / BK;
    for (int kt = 0; kt < nk; ++kt) {
#pragma unroll
        for (int i = 0; i < 8; ++i) {       // A tile: 256x32 f32 via float4
            const int e = tid + i * 256;    // float4 index 0..2047
            const int r = e >> 3;
            const int c = (e & 7) << 2;
            const float4 va = *(const float4*)(A + (size_t)(brow + r) * K + (size_t)kt * BK + c);
            As[r][c + 0] = (bf16_t)va.x; As[r][c + 1] = (bf16_t)va.y;
            As[r][c + 2] = (bf16_t)va.z; As[r][c + 3] = (bf16_t)va.w;
        }
#pragma unroll
        for (int i = 0; i < 4; ++i) {       // B tile: 32x128 f32, store transposed
            const int e = tid + i * 256;
            const int r = e >> 5;
            const int c = (e & 31) << 2;
            const float4 vb = *(const float4*)(B + (size_t)(kt * BK + r) * N + bcol + c);
            Bt[c + 0][r] = (bf16_t)vb.x; Bt[c + 1][r] = (bf16_t)vb.y;
            Bt[c + 2][r] = (bf16_t)vb.z; Bt[c + 3][r] = (bf16_t)vb.w;
        }
        __syncthreads();

        v16bf bq[4];
#pragma unroll
        for (int ni = 0; ni < 4; ++ni) {    // 16 contiguous bf16 -> 2x b128
            const int n = wn * 64 + ni * 16 + ml;
#pragma unroll
            for (int i = 0; i < 16; ++i) bq[ni][i] = Bt[n][kb2 + i];
        }
#pragma unroll
        for (int mi = 0; mi < 4; ++mi) {
            v16bf af;                        // two 16B runs -> 2x b128
            const int m = wm * 64 + mi * 16 + ml;
#pragma unroll
            for (int i = 0; i < 8; ++i) {
                af[i]     = As[m][kb2 + i];
                af[8 + i] = As[m][kb2 + 8 + i];
            }
#pragma unroll
            for (int ni = 0; ni < 4; ++ni)
                acc[mi][ni] = __builtin_amdgcn_wmma_f32_16x16x32_bf16(
                    false, af, false, bq[ni], (short)0, acc[mi][ni], false, false);
        }
        __syncthreads();
    }

    const int rh = hh * 8;
#pragma unroll
    for (int mi = 0; mi < 4; ++mi) {
#pragma unroll
        for (int ni = 0; ni < 4; ++ni) {
            const int col = bcol + wn * 64 + ni * 16 + ml;
            const float bv = bias[col];
#pragma unroll
            for (int r = 0; r < 8; ++r) {
                const int row = brow + wm * 64 + mi * 16 + rh + r;
                float v = acc[mi][ni][r] + bv;
                if (EP == 2) v = gelu_tanh(v);
                if (EP == 3) v = resid[(size_t)row * N + col] + v * gate[col];
                C[(size_t)row * N + col] = v;
            }
        }
    }
}

// ---------------------------------------------------------------------------
// RMS-norm q/k per head (128), RoPE on image rows, pre-scale q by 1/sqrt(hd)
// ---------------------------------------------------------------------------
__global__ __launch_bounds__(128) void qkv_prep(const float* __restrict__ hbuf,
                                                const float* __restrict__ qnw,
                                                const float* __restrict__ knw,
                                                const float* __restrict__ fcos,
                                                const float* __restrict__ fsin,
                                                float* __restrict__ Q,
                                                float* __restrict__ Kb,
                                                float* __restrict__ Vb) {
    const int l = blockIdx.x, h = blockIdx.y, d = threadIdx.x;
    __shared__ float red[128];
    __shared__ float bq[128], bk[128];
    const size_t base = (size_t)l * N1;
    const float qv = hbuf[base + h * HD + d];
    const float kv = hbuf[base + D_MODEL + h * HD + d];
    const float vv = hbuf[base + 2 * D_MODEL + h * HD + d];
    const float sq = blockReduceSum<128>(qv * qv, red);
    float qn = qv * rsqrtf(sq * (1.0f / HD) + LN_EPS) * qnw[d];
    const float sk = blockReduceSum<128>(kv * kv, red);
    float kn = kv * rsqrtf(sk * (1.0f / HD) + LN_EPS) * knw[d];
    bq[d] = qn;
    bk[d] = kn;
    __syncthreads();
    if (l < IMG) {
        const int p = d >> 1;
        const float c = fcos[l * (HD / 2) + p];
        const float s = fsin[l * (HD / 2) + p];
        const float qe = bq[p * 2], qo = bq[p * 2 + 1];
        const float ke = bk[p * 2], ko = bk[p * 2 + 1];
        qn = (d & 1) ? (qe * s + qo * c) : (qe * c - qo * s);
        kn = (d & 1) ? (ke * s + ko * c) : (ke * c - ko * s);
    }
    Q[(size_t)l * D_MODEL + h * HD + d]  = qn * QK_SCALE;
    Kb[(size_t)l * D_MODEL + h * HD + d] = kn;
    Vb[(size_t)l * D_MODEL + h * HD + d] = vv;
}

// ---------------------------------------------------------------------------
// Flash attention: block = (query tile of 128, head); 8 waves x 16 rows each.
// K-tile kept natural [s][d]; V-tile transposed [d][s] so all WMMA B-fragments
// load as contiguous 2x ds_load_b128. Writes into acat (ref rows dropped).
// ---------------------------------------------------------------------------
__global__ __launch_bounds__(256) void flash_attn(const float* __restrict__ Q,
                                                  const float* __restrict__ Kb,
                                                  const float* __restrict__ Vb,
                                                  const int* __restrict__ tmask,
                                                  float* __restrict__ Ocat) {
    const int qt = blockIdx.x;   // 0..21
    const int h  = blockIdx.y;   // 0..23
    if (qt >= IMG / 128 && qt < (IMG + REFN) / 128) return;  // pure-ref tiles discarded

    __shared__ bf16_t Ks[KT][HD + 8];        // [key pos][head-dim]
    __shared__ bf16_t Vt[HD][KT + 8];        // [head-dim][key pos] (transposed)
    __shared__ bf16_t Ps[8][16][KT + 8];     // per-wave P tile

    const int tid  = threadIdx.x;
    const int lane = tid & 31;
    const int wid  = tid >> 5;
    const int ml   = lane & 15;
    const int hh   = lane >> 4;
    const int kb   = hh * 8;
    const int kb2  = hh * 16;
    const int rh   = hh * 8;

    // Q fragments (A layout), 4 k-steps over head dim
    const int qrow = qt * 128 + wid * 16 + ml;
    v16bf qf[4];
#pragma unroll
    for (int ks = 0; ks < 4; ++ks) {
        const float* qp = Q + (size_t)qrow * D_MODEL + h * HD + ks * 32;
#pragma unroll
        for (int i = 0; i < 8; ++i) {
            qf[ks][i]     = (bf16_t)qp[kb2 + i];
            qf[ks][8 + i] = (bf16_t)qp[kb2 + 8 + i];
        }
    }

    v8f oacc[8];
#pragma unroll
    for (int i = 0; i < 8; ++i)
#pragma unroll
        for (int e = 0; e < 8; ++e) oacc[i][e] = 0.0f;
    float mrow[8], lrow[8];
#pragma unroll
    for (int r = 0; r < 8; ++r) { mrow[r] = -3.0e38f; lrow[r] = 0.0f; }

    for (int kt = 0; kt < LTOT / KT; ++kt) {
        // cooperative load K (natural) and V (transposed): KT x 128 floats each
#pragma unroll
        for (int i = 0; i < 8; ++i) {
            const int e = tid + i * 256;   // float4 index 0..2047
            const int s = e >> 5;
            const int d = (e & 31) << 2;
            const float4 kv = *(const float4*)(Kb + (size_t)(kt * KT + s) * D_MODEL + h * HD + d);
            Ks[s][d + 0] = (bf16_t)kv.x; Ks[s][d + 1] = (bf16_t)kv.y;
            Ks[s][d + 2] = (bf16_t)kv.z; Ks[s][d + 3] = (bf16_t)kv.w;
            const float4 vv = *(const float4*)(Vb + (size_t)(kt * KT + s) * D_MODEL + h * HD + d);
            Vt[d + 0][s] = (bf16_t)vv.x; Vt[d + 1][s] = (bf16_t)vv.y;
            Vt[d + 2][s] = (bf16_t)vv.z; Vt[d + 3][s] = (bf16_t)vv.w;
        }
        __syncthreads();

        // S = Q @ K^T  (16 x KT per wave); q already scaled by 1/sqrt(hd)
        v8f sacc[4];
#pragma unroll
        for (int i = 0; i < 4; ++i)
#pragma unroll
            for (int e = 0; e < 8; ++e) sacc[i][e] = 0.0f;
#pragma unroll
        for (int ks = 0; ks < 4; ++ks) {
#pragma unroll
            for (int nt = 0; nt < 4; ++nt) {
                v16bf bfr;
#pragma unroll
                for (int i = 0; i < 16; ++i) bfr[i] = Ks[nt * 16 + ml][ks * 32 + kb2 + i];
                sacc[nt] = __builtin_amdgcn_wmma_f32_16x16x32_bf16(
                    false, qf[ks], false, bfr, (short)0, sacc[nt], false, false);
            }
        }

        // mask + online softmax
        float sv[4][8];
#pragma unroll
        for (int nt = 0; nt < 4; ++nt) {
            const int j = kt * KT + nt * 16 + ml;
            const bool ok = (j < IMG + REFN) || (tmask[j - (IMG + REFN)] != 0);
#pragma unroll
            for (int r = 0; r < 8; ++r) sv[nt][r] = ok ? sacc[nt][r] : -1.0e30f;
        }
        float tmax[8];
#pragma unroll
        for (int r = 0; r < 8; ++r) {
            float t = sv[0][r];
#pragma unroll
            for (int nt = 1; nt < 4; ++nt) t = fmaxf(t, sv[nt][r]);
#pragma unroll
            for (int off = 1; off < 16; off <<= 1) t = fmaxf(t, __shfl_xor(t, off, 32));
            tmax[r] = t;
        }
        float mnew[8], scl[8], psum[8];
#pragma unroll
        for (int r = 0; r < 8; ++r) {
            mnew[r] = fmaxf(mrow[r], tmax[r]);
            scl[r]  = __expf(mrow[r] - mnew[r]);
            mrow[r] = mnew[r];
            psum[r] = 0.0f;
        }
#pragma unroll
        for (int nt = 0; nt < 4; ++nt)
#pragma unroll
            for (int r = 0; r < 8; ++r) {
                const float p = __expf(sv[nt][r] - mnew[r]);
                psum[r] += p;
                Ps[wid][rh + r][nt * 16 + ml] = (bf16_t)p;
            }
#pragma unroll
        for (int r = 0; r < 8; ++r) {
            float t = psum[r];
#pragma unroll
            for (int off = 1; off < 16; off <<= 1) t += __shfl_xor(t, off, 32);
            lrow[r] = lrow[r] * scl[r] + t;
        }
#pragma unroll
        for (int ht = 0; ht < 8; ++ht)
#pragma unroll
            for (int r = 0; r < 8; ++r) oacc[ht][r] *= scl[r];
        __syncthreads();   // Ps visible; all waves done with Ks

        // O += P @ V
#pragma unroll
        for (int ks2 = 0; ks2 < 2; ++ks2) {
            v16bf pf;
#pragma unroll
            for (int i = 0; i < 8; ++i) {
                pf[i]     = Ps[wid][ml][ks2 * 32 + kb2 + i];
                pf[8 + i] = Ps[wid][ml][ks2 * 32 + kb2 + 8 + i];
            }
#pragma unroll
            for (int ht = 0; ht < 8; ++ht) {
                v16bf bfr;
#pragma unroll
                for (int i = 0; i < 16; ++i) bfr[i] = Vt[ht * 16 + ml][ks2 * 32 + kb2 + i];
                oacc[ht] = __builtin_amdgcn_wmma_f32_16x16x32_bf16(
                    false, pf, false, bfr, (short)0, oacc[ht], false, false);
            }
        }
        __syncthreads();   // before next tile overwrites Ks/Vt/Ps
    }

    // normalize and store (drop ref rows)
#pragma unroll
    for (int ht = 0; ht < 8; ++ht) {
        const int col = h * HD + ht * 16 + ml;
#pragma unroll
        for (int r = 0; r < 8; ++r) {
            int row = qt * 128 + wid * 16 + rh + r;
            if (row >= IMG + REFN) row -= REFN;
            else if (row >= IMG) continue;
            Ocat[(size_t)row * N2CAT + col] = oacc[ht][r] / lrow[r];
        }
    }
}

// ---------------------------------------------------------------------------
// gelu(mlp) into concat buffer (ref rows dropped)
// ---------------------------------------------------------------------------
__global__ __launch_bounds__(256) void mlp_gelu(const float* __restrict__ hbuf,
                                                float* __restrict__ acat) {
    const int row = blockIdx.x;
    const int j   = blockIdx.y * 256 + threadIdx.x;
    const int l   = (row < IMG) ? row : row + REFN;
    acat[(size_t)row * N2CAT + D_MODEL + j] =
        gelu_tanh(hbuf[(size_t)l * N1 + 3 * D_MODEL + j]);
}

// ---------------------------------------------------------------------------
extern "C" void kernel_launch(void* const* d_in, const int* in_sizes, int n_in,
                              void* d_out, int out_size, void* d_ws, size_t ws_size,
                              hipStream_t stream) {
    (void)in_sizes; (void)n_in; (void)out_size; (void)ws_size;
    const float* x      = (const float*)d_in[0];
    const float* vec    = (const float*)d_in[1];
    const float* refp   = (const float*)d_in[2];
    const float* fcos   = (const float*)d_in[3];
    const float* fsin   = (const float*)d_in[4];
    const float* mod_w  = (const float*)d_in[5];
    const float* mod_b  = (const float*)d_in[6];
    const float* fc1_w  = (const float*)d_in[7];
    const float* fc1_b  = (const float*)d_in[8];
    const float* fc2_w  = (const float*)d_in[9];
    const float* fc2_b  = (const float*)d_in[10];
    const float* rn_w   = (const float*)d_in[11];
    const float* rn_b   = (const float*)d_in[12];
    const float* lin1_w = (const float*)d_in[13];
    const float* lin1_b = (const float*)d_in[14];
    const float* lin2_w = (const float*)d_in[15];
    const float* lin2_b = (const float*)d_in[16];
    const float* qn_w   = (const float*)d_in[17];
    const float* kn_w   = (const float*)d_in[18];
    const int*   tmask  = (const int*)d_in[19];
    float* out = (float*)d_out;

    float* ws   = (float*)d_ws;
    float* mod  = ws;                                   // 9216
    float* r1   = mod + 9216;                           // 512*3072
    float* xm   = r1 + (size_t)REFN * D_MODEL;          // 2816*3072
    float* hbuf = xm + (size_t)LTOT * D_MODEL;          // 2816*21504
    float* qb   = hbuf + (size_t)LTOT * N1;             // 2816*3072
    float* kb   = qb + (size_t)LTOT * D_MODEL;
    float* vb   = kb + (size_t)LTOT * D_MODEL;
    float* acat = vb + (size_t)LTOT * D_MODEL;          // 2304*15360

    mod_gemv<<<3 * D_MODEL / 256, 256, 0, stream>>>(vec, mod_w, mod_b, mod);
    ln_mod_x<<<XLEN, 256, 0, stream>>>(x, mod, xm);
    gemm_bf16<2><<<dim3(D_MODEL / BN, REFN / BM), 256, 0, stream>>>(
        refp, fc1_w, fc1_b, r1, REFN, D_MODEL, D_MODEL, nullptr, nullptr);
    gemm_bf16<1><<<dim3(D_MODEL / BN, REFN / BM), 256, 0, stream>>>(
        r1, fc2_w, fc2_b, xm + (size_t)IMG * D_MODEL, REFN, D_MODEL, D_MODEL, nullptr, nullptr);
    ln_affine<<<REFN, 256, 0, stream>>>(xm + (size_t)IMG * D_MODEL, rn_w, rn_b);
    gemm_bf16<1><<<dim3(N1 / BN, LTOT / BM), 256, 0, stream>>>(
        xm, lin1_w, lin1_b, hbuf, LTOT, N1, D_MODEL, nullptr, nullptr);
    qkv_prep<<<dim3(LTOT, NHEAD), 128, 0, stream>>>(hbuf, qn_w, kn_w, fcos, fsin, qb, kb, vb);
    flash_attn<<<dim3(LTOT / 128, NHEAD), 256, 0, stream>>>(qb, kb, vb, tmask, acat);
    mlp_gelu<<<dim3(XLEN, MLPD / 256), 256, 0, stream>>>(hbuf, acat);
    gemm_bf16<3><<<dim3(D_MODEL / BN, XLEN / BM), 256, 0, stream>>>(
        acat, lin2_w, lin2_b, out, XLEN, D_MODEL, N2CAT, x, mod + 2 * D_MODEL);
}
